// KMISPooling_15470472201062
// MI455X (gfx1250) — matrix operand order repaired and checked by reading, not compiled, not run on hardware
//
#include <hip/hip_runtime.h>
#include <hip/hip_bf16.h>
#include <climits>
#include <cstdint>

typedef __attribute__((ext_vector_type(16))) _Float16 v16h;
typedef __attribute__((ext_vector_type(8)))  float    v8f;

#define MAX_IT 64

// ---------------------------------------------------------------------------
// 0. zero d_out
// ---------------------------------------------------------------------------
__global__ void k_zero_out(float* out, long n) {
  long i = (long)blockIdx.x * blockDim.x + threadIdx.x;
  if (i < n) out[i] = 0.0f;
}

// ---------------------------------------------------------------------------
// 1. init workspace state (must run every call: ws is not re-poisoned)
// ---------------------------------------------------------------------------
__global__ void k_init(float* deg, int* mis, int* neigh, int* counters, int* conv,
                       float* eattr, int n, int E) {
  int i = blockIdx.x * blockDim.x + threadIdx.x;
  if (i < n) { deg[i] = 0.0f; mis[i] = 0; neigh[i] = n; }
  if (i < E) eattr[i] = 0.0f;
  if (i < MAX_IT) counters[i] = 0;
  if (i == 0) *conv = 0;
}

// ---------------------------------------------------------------------------
// 2. scorer: score = sigmoid(F @ w + b) via v_wmma_f32_16x16x32_f16.
//    One wave32 computes a 16-row tile. A = 16x32 f16 feature tile; per the
//    ISA 7.12.2 layout each lane's 16 K-elements are two contiguous 8-float
//    runs -> four global_load_b128 per K-step. B = 32x16 with column N=0
//    carrying w[k..k+31]; w is staged into LDS with the CDNA5 async path
//    (global_load_async_to_lds_b32 + s_wait_asynccnt) and read as float4.
//    C/D f32 layout: N=0 lives in lane 0 (M=0..7) and lane 16 (M=8..15).
// ---------------------------------------------------------------------------
template <int DD>
__global__ void k_score_wmma(const float* __restrict__ F, const float* __restrict__ w,
                             const float* __restrict__ b, float* __restrict__ score,
                             int n) {
  __shared__ __align__(16) float wl[DD];
  int tid = threadIdx.x;

  // async-stage w[0..DD) into LDS (ASYNCcnt-tracked)
  for (int i = tid; i < DD; i += blockDim.x) {
    unsigned lds_off = (unsigned)(uintptr_t)(&wl[i]);
    const float* gp = w + i;
    asm volatile("global_load_async_to_lds_b32 %0, %1, off"
                 :: "v"(lds_off), "v"(gp) : "memory");
  }
  asm volatile("s_wait_asynccnt 0x0" ::: "memory");
  __syncthreads();

  int wave = (blockIdx.x * blockDim.x + tid) >> 5;  // wave32
  int lane = tid & 31;
  int row0 = wave * 16;
  if (row0 >= n) return;               // uniform per wave; no barriers below
  int m  = lane & 15;
  int hi = lane >> 4;                  // 0 or 1
  int row = row0 + m;
  if (row >= n) row = n - 1;           // clamp: loads valid, stores guarded
  const float4* frow = (const float4*)(F + (size_t)row * DD);
  float msk = (m == 0) ? 1.0f : 0.0f;

  v8f acc = {};
#pragma unroll
  for (int k = 0; k < DD; k += 32) {
    // A: contiguous runs [k+hi*8 .. +7] and [k+16+hi*8 .. +7] (16B aligned)
    int c0 = (k + hi * 8) >> 2;
    int c1 = (k + 16 + hi * 8) >> 2;
    float4 a0 = frow[c0], a1 = frow[c0 + 1];
    float4 a2 = frow[c1], a3 = frow[c1 + 1];
    v16h a;
    a[0]  = (_Float16)a0.x; a[1]  = (_Float16)a0.y;
    a[2]  = (_Float16)a0.z; a[3]  = (_Float16)a0.w;
    a[4]  = (_Float16)a1.x; a[5]  = (_Float16)a1.y;
    a[6]  = (_Float16)a1.z; a[7]  = (_Float16)a1.w;
    a[8]  = (_Float16)a2.x; a[9]  = (_Float16)a2.y;
    a[10] = (_Float16)a2.z; a[11] = (_Float16)a2.w;
    a[12] = (_Float16)a3.x; a[13] = (_Float16)a3.y;
    a[14] = (_Float16)a3.z; a[15] = (_Float16)a3.w;

    // B: lane holds K = hi*8 + j (j<8), 16 + hi*8 + (j-8) (j>=8), only column N==0 (m==0) nonzero
    const float4* wp = (const float4*)&wl[k + hi * 8];
    const float4* wq = (const float4*)&wl[k + 16 + hi * 8];
    float4 w0 = wp[0], w1 = wp[1], w2 = wq[0], w3 = wq[1];
    v16h bm;
    bm[0]  = (_Float16)(w0.x * msk); bm[1]  = (_Float16)(w0.y * msk);
    bm[2]  = (_Float16)(w0.z * msk); bm[3]  = (_Float16)(w0.w * msk);
    bm[4]  = (_Float16)(w1.x * msk); bm[5]  = (_Float16)(w1.y * msk);
    bm[6]  = (_Float16)(w1.z * msk); bm[7]  = (_Float16)(w1.w * msk);
    bm[8]  = (_Float16)(w2.x * msk); bm[9]  = (_Float16)(w2.y * msk);
    bm[10] = (_Float16)(w2.z * msk); bm[11] = (_Float16)(w2.w * msk);
    bm[12] = (_Float16)(w3.x * msk); bm[13] = (_Float16)(w3.y * msk);
    bm[14] = (_Float16)(w3.z * msk); bm[15] = (_Float16)(w3.w * msk);

    acc = __builtin_amdgcn_wmma_f32_16x16x32_f16(false, a, false, bm,
                                                 (short)0, acc, false, false);
  }

  if (m == 0) {
#pragma unroll
    for (int r = 0; r < 8; ++r) {
      int orow = row0 + hi * 8 + r;
      if (orow < n) {
        float s = acc[r] + b[0];
        score[orow] = 1.0f / (1.0f + expf(-s));
      }
    }
  }
}

// generic fallback (D != 256): plain dot product
__global__ void k_score_generic(const float* __restrict__ F, const float* __restrict__ w,
                                const float* __restrict__ b, float* __restrict__ score,
                                int n, int d) {
  int i = blockIdx.x * blockDim.x + threadIdx.x;
  if (i >= n) return;
  float s = b[0];
  for (int k = 0; k < d; ++k) s += F[(size_t)i * d + k] * w[k];
  score[i] = 1.0f / (1.0f + expf(-s));
}

// ---------------------------------------------------------------------------
// 3. k_sums (K=1): deg[c] = in-degree; updated = score/(1+deg); fill sort keys
// ---------------------------------------------------------------------------
__global__ void k_deg(const long long* __restrict__ col, float* deg, int E) {
  int e = blockIdx.x * blockDim.x + threadIdx.x;
  if (e < E) atomicAdd(&deg[(int)col[e]], 1.0f);
}

__global__ void k_updkeys(const float* score, const float* deg,
                          float* skey, int* sval, int n, int pn) {
  int i = blockIdx.x * blockDim.x + threadIdx.x;
  if (i >= pn) return;
  if (i < n) { skey[i] = score[i] / (1.0f + deg[i]); sval[i] = i; }
  else       { skey[i] = -__builtin_inff();          sval[i] = INT_MAX; }
}

// bitonic stage, float keys: final order = (key desc, idx asc)  [stable argsort(-x)]
__global__ void k_sort_node(float* key, int* val, int len, int j, int k) {
  int i = blockIdx.x * blockDim.x + threadIdx.x;
  if (i >= len) return;
  int ixj = i ^ j;
  if (ixj <= i || ixj >= len) return;
  float ka = key[i], kb = key[ixj];
  int va = val[i], vb = val[ixj];
  bool ab = (ka > kb) || (ka == kb && va < vb);   // a precedes b
  bool up = ((i & k) == 0);
  if (up ? !ab : ab) { key[i] = kb; key[ixj] = ka; val[i] = vb; val[ixj] = va; }
}

__global__ void k_permrank(const int* sval, int* perm, int* rank, int n) {
  int i = blockIdx.x * blockDim.x + threadIdx.x;
  if (i >= n) return;
  int node = sval[i];
  perm[i] = node;
  rank[node] = i;
}

__global__ void k_minrank_init(const int* rank, int* minrank, int n) {
  int i = blockIdx.x * blockDim.x + threadIdx.x;
  if (i < n) minrank[i] = rank[i];
}

// ---------------------------------------------------------------------------
// 4. MIS fixed-point loop (bounded; early-out on device flag)
// ---------------------------------------------------------------------------
__global__ void k_edge_min(const int* conv, const long long* row, const long long* col,
                           const int* minrank, int* neigh, int E) {
  if (*conv) return;
  int e = blockIdx.x * blockDim.x + threadIdx.x;
  if (e < E) atomicMin(&neigh[(int)col[e]], minrank[(int)row[e]]);
}

__global__ void k_node_a(const int* conv, const int* rank, int* minrank, int* neigh,
                         int* mis, int* marr, int* m2, int n) {
  if (*conv) return;
  int i = blockIdx.x * blockDim.x + threadIdx.x;
  if (i >= n) return;
  int mr = min(minrank[i], neigh[i]);
  minrank[i] = mr;
  neigh[i] = n;                         // reset for next propagation
  int ms = mis[i] | (rank[i] == mr);
  mis[i] = ms; marr[i] = ms; m2[i] = ms;
}

__global__ void k_edge_max(const int* conv, const long long* row, const long long* col,
                           const int* marr, int* m2, int E) {
  if (*conv) return;
  int e = blockIdx.x * blockDim.x + threadIdx.x;
  if (e < E) atomicMax(&m2[(int)col[e]], marr[(int)row[e]]);
}

__global__ void k_node_b(const int* conv, const int* rank, const int* marr, const int* m2,
                         int* minrank, int* counters, int it, int n) {
  if (*conv) return;
  int i = blockIdx.x * blockDim.x + threadIdx.x;
  if (i >= n) return;
  int mm = max(marr[i], m2[i]);         // mask
  if (!mm) atomicAdd(&counters[it], 1);
  minrank[i] = mm ? n : rank[i];
}

__global__ void k_check(const int* counters, int it, int* conv) {
  if (blockIdx.x == 0 && threadIdx.x == 0)
    if (counters[it] == 0) *conv = 1;
}

// ---------------------------------------------------------------------------
// 5. clustering: propagate nearest MIS rank, then counting-based unique
// ---------------------------------------------------------------------------
__global__ void k_cluster_init(const int* mis, const int* rank, int* minrank2, int n) {
  int i = blockIdx.x * blockDim.x + threadIdx.x;
  if (i < n) minrank2[i] = mis[i] ? rank[i] : n;
}

__global__ void k_edge_min2(const long long* row, const long long* col,
                            const int* minrank2, int* neigh, int E) {
  int e = blockIdx.x * blockDim.x + threadIdx.x;
  if (e < E) atomicMin(&neigh[(int)col[e]], minrank2[(int)row[e]]);
}

__global__ void k_node_min2(int* minrank2, const int* neigh, int n) {
  int i = blockIdx.x * blockDim.x + threadIdx.x;
  if (i < n) minrank2[i] = min(minrank2[i], neigh[i]);
}

// single-workgroup exclusive scan (LDS-based; no timing pressure)
__global__ void k_scan(const int* __restrict__ in, int* __restrict__ excl,
                       int* __restrict__ total, int n) {
  __shared__ int sm[1024];
  __shared__ int carry;
  int tid = threadIdx.x;
  if (tid == 0) carry = 0;
  __syncthreads();
  for (int base = 0; base < n; base += 1024) {
    int i = base + tid;
    int v = (i < n) ? in[i] : 0;
    sm[tid] = v;
    __syncthreads();
    for (int off = 1; off < 1024; off <<= 1) {
      int add = (tid >= off) ? sm[tid - off] : 0;
      __syncthreads();
      sm[tid] += add;
      __syncthreads();
    }
    if (i < n) excl[i] = carry + sm[tid] - v;
    __syncthreads();
    if (tid == 0) carry += sm[1023];
    __syncthreads();
  }
  if (tid == 0) *total = carry;
}

// cluster[i] = index (among MIS nodes, node order) of MIS node whose rank == minrank2[i]
__global__ void k_cluster_final(const int* minrank2, const int* perm, const int* misIdx,
                                int* cluster, int n) {
  int i = blockIdx.x * blockDim.x + threadIdx.x;
  if (i >= n) return;
  int r = minrank2[i];
  int node = (r < n) ? perm[r] : i;
  cluster[i] = misIdx[node];
}

// ---------------------------------------------------------------------------
// 6. pooled-adjacency coalesce: 64-bit keys, bitonic sort, head-flag + scan
// ---------------------------------------------------------------------------
__global__ void k_ekeys(const long long* row, const long long* col, const float* val,
                        const int* cluster, const int* cCount,
                        long long* ekey, float* eval, int E, int pe) {
  int e = blockIdx.x * blockDim.x + threadIdx.x;
  if (e >= pe) return;
  if (e < E) {
    long long c = (long long)max(*cCount, 1);
    ekey[e] = (long long)cluster[(int)row[e]] * c + (long long)cluster[(int)col[e]];
    eval[e] = val[e];
  } else {
    ekey[e] = LLONG_MAX;
    eval[e] = 0.0f;
  }
}

__global__ void k_sort_edge(long long* key, float* val, int len, int j, int k) {
  int i = blockIdx.x * blockDim.x + threadIdx.x;
  if (i >= len) return;
  int ixj = i ^ j;
  if (ixj <= i || ixj >= len) return;
  long long ka = key[i], kb = key[ixj];
  float va = val[i], vb = val[ixj];
  bool ab = (ka < kb);                  // ascending; tie order irrelevant (sum)
  bool up = ((i & k) == 0);
  if (up ? !ab : ab) { key[i] = kb; key[ixj] = ka; val[i] = vb; val[ixj] = va; }
}

__global__ void k_heads(const long long* ekey, int* eflag, int E) {
  int e = blockIdx.x * blockDim.x + threadIdx.x;
  if (e < E) eflag[e] = (e == 0) || (ekey[e] != ekey[e - 1]);
}

__global__ void k_accum(const long long* ekey, const float* eval, const int* eflag,
                        const int* eidx, const int* cCount,
                        int* er, int* ec, float* eattr, int E) {
  int e = blockIdx.x * blockDim.x + threadIdx.x;
  if (e >= E) return;
  int g = eidx[e] + eflag[e] - 1;
  atomicAdd(&eattr[g], eval[e]);
  if (eflag[e]) {
    long long c = (long long)max(*cCount, 1);
    long long key = ekey[e];
    er[g] = (int)(key / c);
    ec[g] = (int)(key % c);
  }
}

// ---------------------------------------------------------------------------
// 7. output writers (offsets computed from device-side c, u)
//    layout: x[c*D] | ei[2u] | ea[u] | batch[c] | mis[n] | cluster[n] | perm_sel[c]
// ---------------------------------------------------------------------------
__global__ void k_write_x(const float* F, const float* score, const int* mis,
                          const int* misIdx, float* out, int n, int D, long out_size) {
  long id = (long)blockIdx.x * blockDim.x + threadIdx.x;
  if (id >= (long)n * D) return;
  int i = (int)(id / D), d = (int)(id % D);
  if (!mis[i]) return;
  long o = (long)misIdx[i] * D + d;
  if (o < out_size) out[o] = F[(long)i * D + d] * score[i];
}

__global__ void k_write_edges(const int* er, const int* ec, const float* eattr,
                              const int* cCount, const int* uCount,
                              float* out, int D, int E, long out_size) {
  int e = blockIdx.x * blockDim.x + threadIdx.x;
  if (e >= E) return;
  long u = *uCount, c = *cCount;
  if (e >= u) return;
  long off = c * D;
  if (off + e < out_size)         out[off + e]         = (float)er[e];
  if (off + u + e < out_size)     out[off + u + e]     = (float)ec[e];
  if (off + 2 * u + e < out_size) out[off + 2 * u + e] = eattr[e];
}

__global__ void k_write_rest(const int* mis, const int* misIdx, const int* cluster,
                             const int* perm, const long long* batch,
                             const int* cCount, const int* uCount,
                             float* out, int n, int D, long out_size) {
  int i = blockIdx.x * blockDim.x + threadIdx.x;
  if (i >= n) return;
  long c = *cCount, u = *uCount;
  long off_batch = c * D + 3 * u;
  long off_mis   = off_batch + c;
  long off_cl    = off_mis + n;
  long off_perm  = off_cl + n;
  if (off_mis + i < out_size) out[off_mis + i] = mis[i] ? 1.0f : 0.0f;
  if (off_cl  + i < out_size) out[off_cl  + i] = (float)cluster[i];
  if (mis[i]) {
    long r = misIdx[i];
    if (off_batch + r < out_size) out[off_batch + r] = (float)batch[i];
    if (off_perm  + r < out_size) out[off_perm  + r] = (float)perm[i];
  }
}

// ---------------------------------------------------------------------------
// host driver
// ---------------------------------------------------------------------------
static inline int npow2(int x) { int p = 1; while (p < x) p <<= 1; return p; }

extern "C" void kernel_launch(void* const* d_in, const int* in_sizes, int n_in,
                              void* d_out, int out_size, void* d_ws, size_t ws_size,
                              hipStream_t stream) {
  const float*     F     = (const float*)d_in[0];
  const long long* eidx0 = (const long long*)d_in[1];
  const float*     eval0 = (const float*)d_in[2];
  const long long* batch = (const long long*)d_in[3];
  const float*     w     = (const float*)d_in[4];
  const float*     b     = (const float*)d_in[5];
  float* out = (float*)d_out;

  const int D = in_sizes[4];                 // 256
  const int n = in_sizes[0] / D;             // 100000
  const int E = in_sizes[2];                 // 3200000
  const long long* erow = eidx0;
  const long long* ecol = eidx0 + E;
  const int pn = npow2(n);
  const int pe = npow2(E);

  // workspace carve (bump allocator, 256B aligned)
  char* p = (char*)d_ws;
  auto carve = [&](size_t bytes) -> void* {
    void* r = (void*)p; p += (bytes + 255) & ~(size_t)255; return r;
  };
  float*     score    = (float*)carve((size_t)n * 4);
  float*     deg      = (float*)carve((size_t)n * 4);
  float*     skey     = (float*)carve((size_t)pn * 4);
  int*       sval     = (int*)carve((size_t)pn * 4);
  int*       perm     = (int*)carve((size_t)n * 4);
  int*       rank     = (int*)carve((size_t)n * 4);
  int*       minrank  = (int*)carve((size_t)n * 4);
  int*       neigh    = (int*)carve((size_t)n * 4);
  int*       mis      = (int*)carve((size_t)n * 4);
  int*       marr     = (int*)carve((size_t)n * 4);
  int*       m2       = (int*)carve((size_t)n * 4);
  int*       minrank2 = (int*)carve((size_t)n * 4);
  int*       misIdx   = (int*)carve((size_t)n * 4);
  int*       cluster  = (int*)carve((size_t)n * 4);
  int*       counters = (int*)carve((size_t)MAX_IT * 4);
  int*       conv     = (int*)carve(4);
  int*       cCount   = (int*)carve(4);
  int*       uCount   = (int*)carve(4);
  long long* ekey     = (long long*)carve((size_t)pe * 8);
  float*     evalw    = (float*)carve((size_t)pe * 4);
  int*       eflag    = (int*)carve((size_t)E * 4);
  int*       eexcl    = (int*)carve((size_t)E * 4);
  int*       er       = (int*)carve((size_t)E * 4);
  int*       ec       = (int*)carve((size_t)E * 4);
  float*     eattr    = (float*)carve((size_t)E * 4);
  (void)ws_size;

  const int BS = 256;
  auto G = [&](long work) { return (int)((work + BS - 1) / BS); };

  // 0. zero output, init state
  k_zero_out<<<G(out_size), BS, 0, stream>>>(out, (long)out_size);
  int initN = (n > E ? n : E);
  k_init<<<G(initN), BS, 0, stream>>>(deg, mis, neigh, counters, conv, eattr, n, E);

  // 1. scorer (WMMA, K-loop fully unrolled for D=256)
  if (D == 256) {
    int waves = (n + 15) / 16;
    k_score_wmma<256><<<G((long)waves * 32), BS, 0, stream>>>(F, w, b, score, n);
  } else {
    k_score_generic<<<G(n), BS, 0, stream>>>(F, w, b, score, n, D);
  }

  // 2. degrees, updated scores, sort keys
  k_deg<<<G(E), BS, 0, stream>>>(ecol, deg, E);
  k_updkeys<<<G(pn), BS, 0, stream>>>(score, deg, skey, sval, n, pn);

  // 3. argsort (bitonic) -> perm, rank
  for (int k = 2; k <= pn; k <<= 1)
    for (int j = k >> 1; j > 0; j >>= 1)
      k_sort_node<<<G(pn), BS, 0, stream>>>(skey, sval, pn, j, k);
  k_permrank<<<G(n), BS, 0, stream>>>(sval, perm, rank, n);
  k_minrank_init<<<G(n), BS, 0, stream>>>(rank, minrank, n);

  // 4. MIS loop (bounded; device-side convergence flag)
  for (int it = 0; it < MAX_IT; ++it) {
    k_edge_min<<<G(E), BS, 0, stream>>>(conv, erow, ecol, minrank, neigh, E);
    k_node_a<<<G(n), BS, 0, stream>>>(conv, rank, minrank, neigh, mis, marr, m2, n);
    k_edge_max<<<G(E), BS, 0, stream>>>(conv, erow, ecol, marr, m2, E);
    k_node_b<<<G(n), BS, 0, stream>>>(conv, rank, marr, m2, minrank, counters, it, n);
    k_check<<<1, 32, 0, stream>>>(counters, it, conv);
  }

  // 5. clusters
  k_cluster_init<<<G(n), BS, 0, stream>>>(mis, rank, minrank2, n);
  k_edge_min2<<<G(E), BS, 0, stream>>>(erow, ecol, minrank2, neigh, E);
  k_node_min2<<<G(n), BS, 0, stream>>>(minrank2, neigh, n);
  k_scan<<<1, 1024, 0, stream>>>(mis, misIdx, cCount, n);
  k_cluster_final<<<G(n), BS, 0, stream>>>(minrank2, perm, misIdx, cluster, n);

  // 6. coalesce pooled adjacency
  k_ekeys<<<G(pe), BS, 0, stream>>>(erow, ecol, eval0, cluster, cCount, ekey, evalw, E, pe);
  for (int k = 2; k <= pe; k <<= 1)
    for (int j = k >> 1; j > 0; j >>= 1)
      k_sort_edge<<<G(pe), BS, 0, stream>>>(ekey, evalw, pe, j, k);
  k_heads<<<G(E), BS, 0, stream>>>(ekey, eflag, E);
  k_scan<<<1, 1024, 0, stream>>>(eflag, eexcl, uCount, E);
  k_accum<<<G(E), BS, 0, stream>>>(ekey, evalw, eflag, eexcl, cCount, er, ec, eattr, E);

  // 7. outputs
  k_write_x<<<G((long)n * D), BS, 0, stream>>>(F, score, mis, misIdx, out, n, D, (long)out_size);
  k_write_edges<<<G(E), BS, 0, stream>>>(er, ec, eattr, cCount, uCount, out, D, E, (long)out_size);
  k_write_rest<<<G(n), BS, 0, stream>>>(mis, misIdx, cluster, perm, batch, cCount, uCount,
                                        out, n, D, (long)out_size);
  (void)n_in;
}
